// Head_72507637891886
// MI455X (gfx1250) — matrix operand order repaired and checked by reading, not compiled, not run on hardware
//
#include <hip/hip_runtime.h>
#include <hip/hip_bf16.h>
#include <math.h>

typedef __attribute__((ext_vector_type(16))) __bf16 v16bf;
typedef __attribute__((ext_vector_type(8)))  __bf16 v8bf;
typedef __attribute__((ext_vector_type(8)))  float  v8f;
typedef __attribute__((ext_vector_type(4)))  int    v4i;

#define ROWS_TOTAL     65536     // B*T = 16*4096
#define E_DIM          1024
#define ROWS_PER_BLOCK 128       // 8 waves * 16 rows
#define NBLOCKS        (ROWS_TOTAL / ROWS_PER_BLOCK)   // 512
#define KC             128       // K-chunk staged in LDS
#define KC_PAD         (KC + 8)  // 136 elems -> 68 dwords row stride: conflict-free
#define NCHUNK         (E_DIM / KC)                    // 8

// workspace layout (bytes)
#define WS_W_OFF    0                          // W[32][4] fp32 normalized feature matrix (512 B)
#define WS_WT_OFF   512                        // Wt[64][1024] bf16 transposed [wq|wk] (131072 B)
#define WS_T_OFF    (512 + 131072)             // t[65536] fp32 staged mean-cosh (262144 B)
#define WS_PART_OFF (WS_T_OFF + 262144)        // per-block partial norms, 2 floats each (4096 B)

#define AS1 __attribute__((address_space(1)))
#define AS3 __attribute__((address_space(3)))

#if defined(__has_builtin)
#if __has_builtin(__builtin_amdgcn_global_load_async_to_lds_b128)
#define HAVE_ASYNC_LDS 1
#endif
#endif
#ifndef HAVE_ASYNC_LDS
#define HAVE_ASYNC_LDS 0
#endif

template <int N>
__device__ __forceinline__ void wait_asynccnt() {
#if HAVE_ASYNC_LDS
#if defined(__has_builtin) && __has_builtin(__builtin_amdgcn_s_wait_asynccnt)
  __builtin_amdgcn_s_wait_asynccnt((short)N);
#else
  asm volatile("s_wait_asynccnt %0" ::"n"(N) : "memory");
#endif
#endif
}

// ---------------------------------------------------------------------------
// Kernel 1: normalize random-feature matrix + transpose/convert weights to bf16
// ---------------------------------------------------------------------------
__global__ __launch_bounds__(256) void prep_kernel(
    const float* __restrict__ wq, const float* __restrict__ wk,
    const float* __restrict__ w_raw, char* __restrict__ ws)
{
  __shared__ float red[128];
  int tid = threadIdx.x;
  if (tid < 128) { float v = w_raw[tid]; red[tid] = v * v; }
  __syncthreads();
  for (int s = 64; s > 0; s >>= 1) {
    if (tid < s) red[tid] += red[tid + s];
    __syncthreads();
  }
  float scale = sqrtf(32.0f) / sqrtf(red[0]);   // sqrt(DIM) / ||w||_F

  float* W = (float*)(ws + WS_W_OFF);           // W[n][j] = scale * w_raw[j][n]
  if (tid < 128) {
    int j = tid >> 5, n = tid & 31;             // w_raw is [OMEGA=4][DIM=32]
    W[n * 4 + j] = scale * w_raw[tid];
  }

  __bf16* Wt = (__bf16*)(ws + WS_WT_OFF);       // Wt[n][k]: n<32 -> wq col n, else wk col n-32
  for (int idx = tid; idx < 64 * 1024; idx += 256) {
    int n = idx >> 10, k = idx & 1023;
    float v = (n < 32) ? wq[k * 32 + n] : wk[k * 32 + (n - 32)];
    Wt[idx] = (__bf16)v;
  }
}

// ---------------------------------------------------------------------------
// Kernel 2: fused q/k projection GEMM (WMMA bf16 -> f32), async-LDS-staged B,
//           norms, z@w, cosh-mean
// ---------------------------------------------------------------------------
__global__ __launch_bounds__(256) void qk_gemm_kernel(
    const float* __restrict__ x, const float* __restrict__ bq,
    const float* __restrict__ bk, char* __restrict__ ws)
{
  const float*  W    = (const float*)(ws + WS_W_OFF);
  const __bf16* Wt   = (const __bf16*)(ws + WS_WT_OFF);
  float*        tOut = (float*)(ws + WS_T_OFF);
  float*        part = (float*)(ws + WS_PART_OFF);

  __shared__ __bf16 WtL[2][64][KC_PAD];   // double-buffered B tile, padded rows

  const int tid    = threadIdx.x;
  const int wave   = tid >> 5;
  const int lane   = tid & 31;
  const int laneLo = lane & 15;
  const int hh     = lane >> 4;                 // half-wave index
  const int rowBase = blockIdx.x * ROWS_PER_BLOCK + wave * 16;

  // A-fragment row handled by this lane (both half-waves cover M=0..15)
  const float* xrow = x + (size_t)(rowBase + laneLo) * E_DIM;

  // Stage one K-chunk of Wt into LDS buffer (chunk & 1).
  // 64 rows x (KC/8)=16 16B-chunks = 1024 chunks, 4 per thread.
  auto stage = [&](int chunk) {
    const int buf = chunk & 1;
    const int kgBase = chunk * KC;
#pragma unroll
    for (int i = 0; i < 4; ++i) {
      int cid = tid + i * 256;
      int n   = cid >> 4;
      int k8  = (cid & 15) * 8;
      const __bf16* src = Wt + (size_t)n * E_DIM + kgBase + k8;
      __bf16*       dst = &WtL[buf][n][k8];
#if HAVE_ASYNC_LDS
      __builtin_amdgcn_global_load_async_to_lds_b128(
          (AS1 v4i*)src, (AS3 v4i*)dst, 0, 0);
#else
      *(v8bf*)dst = *(const v8bf*)src;
#endif
    }
  };

  v8f cq0 = {}, cq1 = {}, ck0 = {}, ck1 = {};

  stage(0);
  for (int c = 0; c < NCHUNK; ++c) {
    const int buf = c & 1;
    if (c + 1 < NCHUNK) {
      stage(c + 1);
      wait_asynccnt<4>();     // in-order completion: chunk c's 4 asyncs landed
    } else {
      wait_asynccnt<0>();
    }
    __syncthreads();          // all waves' staging visible

#pragma unroll
    for (int k0l = 0; k0l < KC; k0l += 32) {
      // ---- A fragment (16-bit A 16x32): lane<16: K 0-7 & 16-23 ; lane>=16: K 8-15 & 24-31
      const int kA = c * KC + k0l + hh * 8;
      float4 a0 = *(const float4*)(xrow + kA);
      float4 a1 = *(const float4*)(xrow + kA + 4);
      float4 a2 = *(const float4*)(xrow + kA + 16);
      float4 a3 = *(const float4*)(xrow + kA + 20);
      v16bf afrag;
      afrag[0]  = (__bf16)a0.x; afrag[1]  = (__bf16)a0.y; afrag[2]  = (__bf16)a0.z; afrag[3]  = (__bf16)a0.w;
      afrag[4]  = (__bf16)a1.x; afrag[5]  = (__bf16)a1.y; afrag[6]  = (__bf16)a1.z; afrag[7]  = (__bf16)a1.w;
      afrag[8]  = (__bf16)a2.x; afrag[9]  = (__bf16)a2.y; afrag[10] = (__bf16)a2.z; afrag[11] = (__bf16)a2.w;
      afrag[12] = (__bf16)a3.x; afrag[13] = (__bf16)a3.y; afrag[14] = (__bf16)a3.z; afrag[15] = (__bf16)a3.w;

      // ---- B fragments (16-bit B 32x16): lane<16: K 0-7 & 16-23 ; lane>=16: K 8-15 & 24-31
      const int kB = k0l + hh * 8;
      auto loadB = [&](int n) -> v16bf {
        const v8bf* p = (const v8bf*)&WtL[buf][n][kB];
        v8bf lo = p[0], hi = p[2];
        v16bf r;
#pragma unroll
        for (int i = 0; i < 8; ++i) { r[i] = lo[i]; r[i + 8] = hi[i]; }
        return r;
      };
      v16bf bfq0 = loadB(laneLo);
      v16bf bfq1 = loadB(laneLo + 16);
      v16bf bfk0 = loadB(laneLo + 32);
      v16bf bfk1 = loadB(laneLo + 48);

      cq0 = __builtin_amdgcn_wmma_f32_16x16x32_bf16(false, afrag, false, bfq0, (short)0, cq0, false, false);
      cq1 = __builtin_amdgcn_wmma_f32_16x16x32_bf16(false, afrag, false, bfq1, (short)0, cq1, false, false);
      ck0 = __builtin_amdgcn_wmma_f32_16x16x32_bf16(false, afrag, false, bfk0, (short)0, ck0, false, false);
      ck1 = __builtin_amdgcn_wmma_f32_16x16x32_bf16(false, afrag, false, bfk1, (short)0, ck1, false, false);
    }
    __syncthreads();          // buffer (c&1) free for reuse by stage(c+2)
  }

  // C/D layout: VGPR r, lane: row = rowBase + r + 8*hh, col = laneLo (+16 for *1 accs)
  const float bq0 = bq[laneLo], bq1 = bq[laneLo + 16];
  const float bk0v = bk[laneLo], bk1v = bk[laneLo + 16];

  float qs = 0.f, ks = 0.f;
  float z0[8], z1[8];
#pragma unroll
  for (int r = 0; r < 8; ++r) {
    float qa = cq0[r] + bq0, qb = cq1[r] + bq1;
    float ka = ck0[r] + bk0v, kb = ck1[r] + bk1v;
    qs += qa * qa + qb * qb;
    ks += ka * ka + kb * kb;
    z0[r] = qa + ka;               // z cols 0..15
    z1[r] = qb + kb;               // z cols 16..31
  }

  // wave-reduce squared norms, then deterministic per-block partial
  for (int off = 16; off >= 1; off >>= 1) {
    qs += __shfl_xor(qs, off, 32);
    ks += __shfl_xor(ks, off, 32);
  }
  __shared__ float lq[8], lk[8];
  if (lane == 0) { lq[wave] = qs; lk[wave] = ks; }
  __syncthreads();
  if (tid == 0) {
    float sq = 0.f, sk = 0.f;
#pragma unroll
    for (int i = 0; i < 8; ++i) { sq += lq[i]; sk += lk[i]; }
    part[2 * blockIdx.x]     = sq;
    part[2 * blockIdx.x + 1] = sk;
  }

  // projection z @ W (32x4) per row: partial per lane, reduce over 16-lane group
  float w0[4], w1[4];
#pragma unroll
  for (int j = 0; j < 4; ++j) {
    w0[j] = W[laneLo * 4 + j];
    w1[j] = W[(laneLo + 16) * 4 + j];
  }
#pragma unroll
  for (int r = 0; r < 8; ++r) {
    float p0 = z0[r] * w0[0] + z1[r] * w1[0];
    float p1 = z0[r] * w0[1] + z1[r] * w1[1];
    float p2 = z0[r] * w0[2] + z1[r] * w1[2];
    float p3 = z0[r] * w0[3] + z1[r] * w1[3];
    for (int off = 1; off <= 8; off <<= 1) {   // reduce within 16-lane half
      p0 += __shfl_xor(p0, off, 32);
      p1 += __shfl_xor(p1, off, 32);
      p2 += __shfl_xor(p2, off, 32);
      p3 += __shfl_xor(p3, off, 32);
    }
    if (laneLo == 0) {
      float mean = 0.25f * (coshf(p0) + coshf(p1) + coshf(p2) + coshf(p3));
      tOut[rowBase + r + 8 * hh] = mean;
    }
  }
}

// ---------------------------------------------------------------------------
// Kernel 3: deterministic reduce of partial norms, scale by exp(-z_sqr/2)
// ---------------------------------------------------------------------------
__global__ __launch_bounds__(256) void finalize_kernel(
    float* __restrict__ out, const char* __restrict__ ws)
{
  const float* part = (const float*)(ws + WS_PART_OFF);
  const float* tOut = (const float*)(ws + WS_T_OFF);
  __shared__ float red[256];
  int tid = threadIdx.x;
  float s = 0.f;
  for (int b = tid; b < NBLOCKS; b += 256) s += part[2 * b] + part[2 * b + 1];
  red[tid] = s;
  __syncthreads();
  for (int st = 128; st > 0; st >>= 1) {
    if (tid < st) red[tid] += red[tid + st];
    __syncthreads();
  }
  float a = expf(-0.5f * red[0]);   // underflows to 0.0f, matching fp32 reference
  for (int i = tid; i < ROWS_TOTAL; i += 256) out[i] = a * tOut[i];
}

// ---------------------------------------------------------------------------
extern "C" void kernel_launch(void* const* d_in, const int* in_sizes, int n_in,
                              void* d_out, int out_size, void* d_ws, size_t ws_size,
                              hipStream_t stream) {
  const float* x     = (const float*)d_in[0];
  const float* wq    = (const float*)d_in[1];
  const float* bq    = (const float*)d_in[2];
  const float* wk    = (const float*)d_in[3];
  const float* bk    = (const float*)d_in[4];
  // d_in[5] = wv, d_in[6] = bv : dead in the reference, never touched
  const float* w_raw = (const float*)d_in[7];
  char*  ws  = (char*)d_ws;
  float* out = (float*)d_out;
  (void)in_sizes; (void)n_in; (void)out_size; (void)ws_size;

  prep_kernel<<<1, 256, 0, stream>>>(wq, wk, w_raw, ws);
  qk_gemm_kernel<<<NBLOCKS, 256, 0, stream>>>(x, bq, bk, ws);
  finalize_kernel<<<1, 256, 0, stream>>>(out, ws);
}
